// GRURNNModel_45088566673416
// MI455X (gfx1250) — matrix operand order repaired and checked by reading, not compiled, not run on hardware
//
#include <hip/hip_runtime.h>
#include <hip/hip_bf16.h>

// ---------------------------------------------------------------------------
// 2-layer GRU (T=512, B=128, IN=64, H=1024) + time-distributed linear (H->64)
// CDNA5 / gfx1250, wave32, WMMA f32 = f16 x f16 (16x16x32), f32 accumulate.
// Weight (B-matrix) tiles staged to LDS via async global->LDS copies
// (ASYNCcnt), double-buffered; latency hidden behind WMMA of current chunk.
// ---------------------------------------------------------------------------

#define T_STEPS 512
#define BATCH   128
#define IN_DIM  64
#define HID     1024
#define OUT_DIM 64

#define KCHUNK  64     // K per LDS stage (2 WMMA k-steps)
#define BROWS   48     // 3 gates x 16 cols
#define BSTRIDE 72     // padded LDS row stride in halves (144B = 36 banks; conflict-free)
#define NCHUNK  (BROWS * (KCHUNK / 8))   // 384 x 16B chunks per stage

#define AS1 __attribute__((address_space(1)))
#define AS3 __attribute__((address_space(3)))

#if defined(__gfx1250__) && __has_builtin(__builtin_amdgcn_global_load_async_to_lds_b128)
#define GRU_ASYNC 1
#else
#define GRU_ASYNC 0
#endif

typedef _Float16 v16h __attribute__((ext_vector_type(16)));
typedef _Float16 v8h  __attribute__((ext_vector_type(8)));
typedef float    v8f  __attribute__((ext_vector_type(8)));
typedef int      v4i  __attribute__((ext_vector_type(4)));

union HV16 { v16h v; v8h h[2]; };

// ---- WMMA fragment loaders (contiguous-K order for A and B, wave32) ------------------
// A (16x32, MxK): lane l(<16) = row m, holds K 0..15; lane l+16 holds
// K 16..31. rowPtr points at A[row] + k0 (row contiguous in K).
__device__ __forceinline__ v16h load_a_frag(const _Float16* __restrict__ rowPtr, int half) {
    HV16 a;
    const int ko = half * 16;
    a.h[0] = *reinterpret_cast<const v8h*>(rowPtr + ko);        // K = ko..ko+7
    a.h[1] = *reinterpret_cast<const v8h*>(rowPtr + ko + 8);    // K = ko+8..ko+15
    return a.v;
}
// B (32x16, KxN) from LDS stage: lane c(<16) = col c, holds K 0..15 (2/VGPR);
// lane c+16 holds K 16..31. kk selects which 32-K slice of the 64-K stage.
__device__ __forceinline__ v16h load_b_lds(const _Float16* __restrict__ buf,
                                           int gate, int lrow, int half, int kk) {
    HV16 b;
    const _Float16* p = buf + (gate * 16 + lrow) * BSTRIDE + kk * 32 + half * 16;
    b.h[0] = *reinterpret_cast<const v8h*>(p);
    b.h[1] = *reinterpret_cast<const v8h*>(p + 8);
    return b.v;
}
// B directly from global (used by the tiny output-linear kernel).
__device__ __forceinline__ v16h load_b_frag(const _Float16* __restrict__ colPtr, int half) {
    HV16 b;
    const int ko = half * 16;
    b.h[0] = *reinterpret_cast<const v8h*>(colPtr + ko);
    b.h[1] = *reinterpret_cast<const v8h*>(colPtr + ko + 8);
    return b.v;
}

__device__ __forceinline__ v8f wmma_f16(v16h a, v16h b, v8f c) {
    return __builtin_amdgcn_wmma_f32_16x16x32_f16(false, a, false, b, (short)0, c, false, false);
}

__device__ __forceinline__ float sigmoidf_fast(float x) {
    return 1.0f / (1.0f + __expf(-x));
}

// ---------------------------------------------------------------------------
// Weight-tile staging: one 64-K x 48-row tile per stage, 384 16B chunks.
// chunk c: row r = c>>3 (gate = r>>4, col = r&15), kp = (c&7)*8 halves.
// Thread tid moves chunk tid; threads 0..127 also move chunk tid+256.
// ---------------------------------------------------------------------------
struct StageRegs { v8h p0; v8h p1; };   // only used by the non-async fallback

__device__ __forceinline__ const _Float16* stage_src(
    const _Float16* __restrict__ W, int ldw, int c0, int k, int c) {
    const int r    = c >> 3;
    const int kp   = (c & 7) * 8;
    const int gate = r >> 4;
    const int col  = r & 15;
    return W + ((size_t)gate * HID + c0 + col) * (size_t)ldw + k + kp;
}
__device__ __forceinline__ _Float16* stage_dst(_Float16* buf, int c) {
    const int r  = c >> 3;
    const int kp = (c & 7) * 8;
    return buf + r * BSTRIDE + kp;
}

__device__ __forceinline__ StageRegs stage_issue(
    const _Float16* __restrict__ W, int ldw, int c0, int k,
    _Float16* buf, int tid) {
    StageRegs sr{};
#if GRU_ASYNC
    // Async DMA: global -> LDS, no VGPR round-trip, tracked by ASYNCcnt.
    __builtin_amdgcn_global_load_async_to_lds_b128(
        (AS1 v4i*)(void*)stage_src(W, ldw, c0, k, tid),
        (AS3 v4i*)stage_dst(buf, tid), 0, 0);
    if (tid < NCHUNK - 256) {
        __builtin_amdgcn_global_load_async_to_lds_b128(
            (AS1 v4i*)(void*)stage_src(W, ldw, c0, k, tid + 256),
            (AS3 v4i*)stage_dst(buf, tid + 256), 0, 0);
    }
#else
    sr.p0 = *reinterpret_cast<const v8h*>(stage_src(W, ldw, c0, k, tid));
    if (tid < NCHUNK - 256)
        sr.p1 = *reinterpret_cast<const v8h*>(stage_src(W, ldw, c0, k, tid + 256));
#endif
    return sr;
}
__device__ __forceinline__ void stage_commit(StageRegs sr, _Float16* buf, int tid) {
#if !GRU_ASYNC
    *reinterpret_cast<v8h*>(stage_dst(buf, tid)) = sr.p0;
    if (tid < NCHUNK - 256)
        *reinterpret_cast<v8h*>(stage_dst(buf, tid + 256)) = sr.p1;
#else
    (void)sr; (void)buf; (void)tid;
#endif
}
__device__ __forceinline__ void stage_fence() {
#if GRU_ASYNC
    asm volatile("s_wait_asynccnt 0" ::: "memory");
#endif
}

// ---------------------------------------------------------------------------
// One GEMM phase: acc{R,Z,3} += A(BATCH x K) @ W_gates^T for this col tile.
// ---------------------------------------------------------------------------
__device__ __forceinline__ void gemm_phase(
    const _Float16* __restrict__ A, int lda,
    const _Float16* __restrict__ W, int ldw, int ktot,
    int c0, int tid, int lane_half, int lrow, int m0,
    _Float16* bs0, _Float16* bs1,
    v8f& accR, v8f& accZ, v8f& acc3) {
    const _Float16* aRow = A + (size_t)(m0 + lrow) * lda;
    _Float16* bufs[2] = { bs0, bs1 };
    const int nst = ktot / KCHUNK;

    StageRegs sr = stage_issue(W, ldw, c0, 0, bufs[0], tid);
    stage_commit(sr, bufs[0], tid);
    stage_fence();
    __syncthreads();

    for (int s = 0; s < nst; ++s) {
        const bool more = (s + 1 < nst);
        StageRegs nx{};
        if (more) nx = stage_issue(W, ldw, c0, (s + 1) * KCHUNK, bufs[(s + 1) & 1], tid);

        const _Float16* buf = bufs[s & 1];
#pragma unroll
        for (int kk = 0; kk < 2; ++kk) {
            v16h a  = load_a_frag(aRow + s * KCHUNK + kk * 32, lane_half);
            v16h fr = load_b_lds(buf, 0, lrow, lane_half, kk);
            v16h fz = load_b_lds(buf, 1, lrow, lane_half, kk);
            v16h fn = load_b_lds(buf, 2, lrow, lane_half, kk);
            accR = wmma_f16(a, fr, accR);
            accZ = wmma_f16(a, fz, accZ);
            acc3 = wmma_f16(a, fn, acc3);
        }

        if (more) stage_commit(nx, bufs[(s + 1) & 1], tid);  // fallback: wait sunk here
        stage_fence();                                       // async: wait hidden behind wmma
        __syncthreads();
    }
}

// ---------------------------------------------------------------------------
// One GRU time step. Grid: HID/16 blocks of 256 threads (8 waves). Block owns
// output cols [c0,c0+16) for all 3 gates; wave w owns batch rows [16w,16w+16).
// ---------------------------------------------------------------------------
__global__ __launch_bounds__(256) void gru_step_kernel(
    const _Float16* __restrict__ xin, int ldx,          // (BATCH, ldx) f16
    const _Float16* __restrict__ hprev,                 // (BATCH, HID) f16
    const _Float16* __restrict__ w_ih,                  // (3H, ldx)   f16
    const _Float16* __restrict__ w_hh,                  // (3H, HID)   f16
    const float*    __restrict__ b_ih,                  // (3H)        f32
    const float*    __restrict__ b_hh,                  // (3H)        f32
    float*          __restrict__ hstate,                // (BATCH,HID) f32 in/out
    _Float16*       __restrict__ hout)                  // (BATCH,HID) f16 out
{
    __shared__ __align__(16) _Float16 bstage[2][BROWS * BSTRIDE];

    const int tid  = threadIdx.x;
    const int lane = tid & 31;
    const int half = lane >> 4;
    const int lrow = lane & 15;
    const int wave = tid >> 5;
    const int m0   = wave * 16;
    const int c0   = blockIdx.x * 16;

    v8f accR  = {};   // r gate: input + hidden combined
    v8f accZ  = {};   // z gate: input + hidden combined
    v8f accIN = {};   // n gate, input side
    v8f accHN = {};   // n gate, hidden side (multiplied by r later)

    // phase 1: input side  x_t @ w_ihT   -> accR, accZ, accIN
    gemm_phase(xin, ldx, w_ih, ldx, ldx, c0, tid, half, lrow, m0,
               bstage[0], bstage[1], accR, accZ, accIN);
    __syncthreads();
    // phase 2: hidden side h_{t-1} @ w_hhT -> accR, accZ, accHN
    gemm_phase(hprev, HID, w_hh, HID, HID, c0, tid, half, lrow, m0,
               bstage[0], bstage[1], accR, accZ, accHN);

    // ---- combine: r,z = sigmoid; n = tanh(in + r*hn); h' = (1-z)n + z h ----
    const int j = c0 + lrow;                   // output column (C layout: N = lane&15)
    const float br  = b_ih[j]           + b_hh[j];
    const float bz  = b_ih[HID + j]     + b_hh[HID + j];
    const float bin = b_ih[2 * HID + j];
    const float bhn = b_hh[2 * HID + j];

#pragma unroll
    for (int v = 0; v < 8; ++v) {
        const int m = m0 + v + half * 8;       // C layout: M = vgpr + 8*(lane>=16)
        const float r  = sigmoidf_fast(accR[v] + br);
        const float z  = sigmoidf_fast(accZ[v] + bz);
        const float n  = tanhf(accIN[v] + bin + r * (accHN[v] + bhn));
        const size_t idx = (size_t)m * HID + j;
        const float hp = hstate[idx];
        const float hv = (1.0f - z) * n + z * hp;
        hstate[idx] = hv;
        hout[idx]   = (_Float16)hv;
    }
}

// ---------------------------------------------------------------------------
// Per-step output linear: out_t = h2_t @ w_linT + b_lin  (128x64, K=1024)
// Grid: OUT_DIM/16 = 4 blocks of 256 threads. Small; direct-global B is fine.
// ---------------------------------------------------------------------------
__global__ __launch_bounds__(256) void linear_step_kernel(
    const _Float16* __restrict__ h2,     // (BATCH, HID) f16
    const _Float16* __restrict__ wlin,   // (OUT, HID)   f16
    const float*    __restrict__ blin,   // (OUT)        f32
    float*          __restrict__ out)    // (BATCH, OUT) f32 (offset to step t)
{
    const int lane = threadIdx.x & 31;
    const int half = lane >> 4;
    const int lrow = lane & 15;
    const int wave = threadIdx.x >> 5;
    const int m0   = wave * 16;
    const int c0   = blockIdx.x * 16;

    v8f acc = {};
    const _Float16* aRow = h2   + (size_t)(m0 + lrow) * HID;
    const _Float16* bCol = wlin + (size_t)(c0 + lrow) * HID;
    for (int k = 0; k < HID; k += 32) {
        v16h a = load_a_frag(aRow + k, half);
        v16h b = load_b_frag(bCol + k, half);
        acc = wmma_f16(a, b, acc);
    }

    const int j = c0 + lrow;
    const float bias = blin[j];
#pragma unroll
    for (int v = 0; v < 8; ++v) {
        const int m = m0 + v + half * 8;
        out[(size_t)m * OUT_DIM + j] = acc[v] + bias;
    }
}

// ---------------------------------------------------------------------------
// Utility kernels
// ---------------------------------------------------------------------------
__global__ void f32_to_f16_kernel(const float* __restrict__ src,
                                  _Float16* __restrict__ dst, int n) {
    for (int i = blockIdx.x * blockDim.x + threadIdx.x; i < n;
         i += gridDim.x * blockDim.x)
        dst[i] = (_Float16)src[i];
}
__global__ void zero_f32_kernel(float* __restrict__ p, int n) {
    for (int i = blockIdx.x * blockDim.x + threadIdx.x; i < n;
         i += gridDim.x * blockDim.x)
        p[i] = 0.0f;
}
__global__ void zero_f16_kernel(_Float16* __restrict__ p, int n) {
    for (int i = blockIdx.x * blockDim.x + threadIdx.x; i < n;
         i += gridDim.x * blockDim.x)
        p[i] = (_Float16)0.0f;
}

// ---------------------------------------------------------------------------
// Host launch
// ---------------------------------------------------------------------------
extern "C" void kernel_launch(void* const* d_in, const int* in_sizes, int n_in,
                              void* d_out, int out_size, void* d_ws, size_t ws_size,
                              hipStream_t stream) {
    const float* x     = (const float*)d_in[0];
    const float* w_ih0 = (const float*)d_in[1];
    const float* w_hh0 = (const float*)d_in[2];
    const float* b_ih0 = (const float*)d_in[3];
    const float* b_hh0 = (const float*)d_in[4];
    const float* w_ih1 = (const float*)d_in[5];
    const float* w_hh1 = (const float*)d_in[6];
    const float* b_ih1 = (const float*)d_in[7];
    const float* b_hh1 = (const float*)d_in[8];
    const float* w_lin = (const float*)d_in[9];
    const float* b_lin = (const float*)d_in[10];
    float* out = (float*)d_out;

    // ---- workspace carve (256B aligned) -----------------------------------
    size_t off = 0;
    auto alloc = [&](size_t bytes) -> void* {
        off = (off + 255) & ~(size_t)255;
        void* p = (char*)d_ws + off;
        off += bytes;
        return p;
    };
    const size_t BH = (size_t)BATCH * HID;
    _Float16* x16    = (_Float16*)alloc((size_t)T_STEPS * BATCH * IN_DIM * 2);
    _Float16* wih0   = (_Float16*)alloc((size_t)3 * HID * IN_DIM * 2);
    _Float16* whh0   = (_Float16*)alloc((size_t)3 * HID * HID * 2);
    _Float16* wih1   = (_Float16*)alloc((size_t)3 * HID * HID * 2);
    _Float16* whh1   = (_Float16*)alloc((size_t)3 * HID * HID * 2);
    _Float16* wlin16 = (_Float16*)alloc((size_t)OUT_DIM * HID * 2);
    _Float16* h1pp   = (_Float16*)alloc(2 * BH * 2);   // layer-0 hidden ping-pong
    _Float16* h2pp   = (_Float16*)alloc(2 * BH * 2);   // layer-1 hidden ping-pong
    _Float16* zero16 = (_Float16*)alloc(BH * 2);       // h_{-1} = 0
    float*    hst0   = (float*)alloc(BH * 4);          // f32 master state, layer 0
    float*    hst1   = (float*)alloc(BH * 4);          // f32 master state, layer 1
    (void)ws_size; (void)n_in; (void)in_sizes; (void)out_size;

    // ---- init + precision conversion --------------------------------------
    zero_f32_kernel<<<256, 256, 0, stream>>>(hst0, (int)BH);
    zero_f32_kernel<<<256, 256, 0, stream>>>(hst1, (int)BH);
    zero_f16_kernel<<<256, 256, 0, stream>>>(zero16, (int)BH);

    f32_to_f16_kernel<<<2048, 256, 0, stream>>>(x, x16, T_STEPS * BATCH * IN_DIM);
    f32_to_f16_kernel<<<256, 256, 0, stream>>>(w_ih0, wih0, 3 * HID * IN_DIM);
    f32_to_f16_kernel<<<2048, 256, 0, stream>>>(w_hh0, whh0, 3 * HID * HID);
    f32_to_f16_kernel<<<2048, 256, 0, stream>>>(w_ih1, wih1, 3 * HID * HID);
    f32_to_f16_kernel<<<2048, 256, 0, stream>>>(w_hh1, whh1, 3 * HID * HID);
    f32_to_f16_kernel<<<256, 256, 0, stream>>>(w_lin, wlin16, OUT_DIM * HID);

    // ---- sequential recurrence --------------------------------------------
    for (int t = 0; t < T_STEPS; ++t) {
        const _Float16* x0     = x16 + (size_t)t * BATCH * IN_DIM;
        const _Float16* h1prev = (t == 0) ? zero16 : h1pp + ((size_t)((t - 1) & 1)) * BH;
        _Float16*       h1out  = h1pp + ((size_t)(t & 1)) * BH;
        gru_step_kernel<<<HID / 16, 256, 0, stream>>>(
            x0, IN_DIM, h1prev, wih0, whh0, b_ih0, b_hh0, hst0, h1out);

        const _Float16* h2prev = (t == 0) ? zero16 : h2pp + ((size_t)((t - 1) & 1)) * BH;
        _Float16*       h2out  = h2pp + ((size_t)(t & 1)) * BH;
        gru_step_kernel<<<HID / 16, 256, 0, stream>>>(
            h1out, HID, h2prev, wih1, whh1, b_ih1, b_hh1, hst1, h2out);

        linear_step_kernel<<<OUT_DIM / 16, 256, 0, stream>>>(
            h2out, wlin16, b_lin, out + (size_t)t * BATCH * OUT_DIM);
    }
}
